// SimpleAttention_13426067767775
// MI455X (gfx1250) — compile-verified, matrix operand order checked by reading, not run on hardware
//
#include <hip/hip_runtime.h>
#include <hip/hip_bf16.h>

typedef __attribute__((ext_vector_type(16))) _Float16 v16h;
typedef __attribute__((ext_vector_type(8)))  _Float16 v8h;
typedef __attribute__((ext_vector_type(8)))  float    v8f;
typedef __attribute__((ext_vector_type(4)))  unsigned int u32x4;
typedef __attribute__((ext_vector_type(8)))  int      i32x8;
typedef __attribute__((ext_vector_type(4)))  int      i32x4;

union H16 { v16h v; v8h h[2]; };

#define WMMA_F16(A, B, C) \
    __builtin_amdgcn_wmma_f32_16x16x32_f16(false, (A), false, (B), (short)0, (C), false, false)

static constexpr int   S_LEN   = 4096;
static constexpr int   DIM     = 256;
static constexpr int   HEADS   = 4;
static constexpr int   HD      = 64;
static constexpr float SCALE   = 0.125f;            // 64^-0.5
static constexpr float LOG2E   = 1.4426950408889634f;
static constexpr float KSCALE  = SCALE * LOG2E;

// ---------------------------------------------------------------------------
// TDM: issue a 2D tensor_load_to_lds.  D# packed per CDNA5 ISA ch.8:
//   group0: count=1 | lds_addr | global_addr(57b) | type=2
//   group1: data_size=1 (2B) | tensor_dim0/1 | tile_dim0/1 | dim0 stride
// Strides/dims are in data_size (element) units.
// This toolchain exposes the 6-arg builtin: (g0, g1, g2, g3, g4, cpol).
// ---------------------------------------------------------------------------
__device__ __forceinline__ void tdm_load_2d(unsigned lds_addr, const void* gaddr,
                                            unsigned tensor_d0, unsigned tensor_d1,
                                            unsigned tile_d0, unsigned tile_d1,
                                            unsigned long long stride0) {
    const unsigned long long ga = (unsigned long long)(size_t)gaddr;
    u32x4 g0;
    g0[0] = 1u;                                                  // count=1, user D#
    g0[1] = lds_addr;                                            // LDS byte address
    g0[2] = (unsigned)(ga & 0xFFFFFFFFu);                        // global_addr[31:0]
    g0[3] = (unsigned)((ga >> 32) & 0x01FFFFFFu) | (2u << 30);   // addr[56:32] | type=2
    i32x8 g1;
    g1[0] = (int)(1u << 16);                                     // data_size=1 (2 bytes)
    g1[1] = (int)((tensor_d0 & 0xFFFFu) << 16);                  // tensor_dim0[15:0]
    g1[2] = (int)((tensor_d0 >> 16) | ((tensor_d1 & 0xFFFFu) << 16));
    g1[3] = (int)((tensor_d1 >> 16) | (tile_d0 << 16));          // | tile_dim0
    g1[4] = (int)(tile_d1 & 0xFFFFu);                            // tile_dim1, tile_dim2=0
    g1[5] = (int)(unsigned)(stride0 & 0xFFFFFFFFu);              // dim0_stride[31:0]
    g1[6] = (int)(unsigned)((stride0 >> 32) & 0xFFFFu);          // dim0_stride[47:32]
    g1[7] = 0;
    i32x4 g2 = {0, 0, 0, 0};
    i32x4 g3 = {0, 0, 0, 0};
    i32x8 g4 = {0, 0, 0, 0, 0, 0, 0, 0};
    __builtin_amdgcn_tensor_load_to_lds(g0, g1, g2, g3, g4, 0);
}

// ---------------------------------------------------------------------------
// Kernel 1: fused QKV projection.  out = x @ W.T  (W stored [out,in]).
// One wave computes one 16x16 output tile via 8x v_wmma_f32_16x16x32_f16.
// Q,K stored f16 as [B*H, S, 64]; V stored transposed as [B*H, 64, S].
// ---------------------------------------------------------------------------
__global__ __launch_bounds__(256) void qkv_proj_kernel(
    const float* __restrict__ x,
    const float* __restrict__ wq, const float* __restrict__ wk, const float* __restrict__ wv,
    _Float16* __restrict__ Q, _Float16* __restrict__ K, _Float16* __restrict__ Vt)
{
    const int w    = blockIdx.x * 8 + (threadIdx.x >> 5);
    const int mat  = w / 16384;              // 0=q 1=k 2=v
    const int rem  = w - mat * 16384;
    const int mb   = (rem >> 4) * 16;        // row tile base in [0, 16384)
    const int nb   = (rem & 15) * 16;        // col tile base in [0, 256)
    const int lane = threadIdx.x & 31;
    const int lo   = lane & 15;
    const int hi   = lane >> 4;

    const float* W = (mat == 0) ? wq : (mat == 1) ? wk : wv;

    v8f acc = {};
    const float* xr = x + (size_t)(mb + lo) * DIM + hi * 8;
    const float* wr = W + (size_t)(nb + lo) * DIM + hi * 8;
#pragma unroll
    for (int k = 0; k < DIM; k += 32) {
        v8f a0 = *(const v8f*)(xr + k);
        v8f a1 = *(const v8f*)(xr + k + 16);
        v8f b0 = *(const v8f*)(wr + k);
        v8f b1 = *(const v8f*)(wr + k + 16);
        H16 af, bf;
#pragma unroll
        for (int i = 0; i < 8; ++i) {
            af.v[i]     = (_Float16)a0[i];
            af.v[i + 8] = (_Float16)a1[i];
            bf.v[i]     = (_Float16)b0[i];
            bf.v[i + 8] = (_Float16)b1[i];
        }
        acc = WMMA_F16(af.v, bf.v, acc);
    }

    const int n = nb + lo;
    const int h = n >> 6;
    const int d = n & 63;
#pragma unroll
    for (int r = 0; r < 8; ++r) {
        const int m  = mb + 8 * hi + r;
        const int b  = m >> 12;
        const int s  = m & 4095;
        const _Float16 val = (_Float16)acc[r];
        const int bh = b * HEADS + h;
        if (mat < 2) {
            _Float16* dst = (mat == 0) ? Q : K;
            dst[((size_t)bh * S_LEN + s) * HD + d] = val;
        } else {
            Vt[((size_t)bh * HD + d) * S_LEN + s] = val;
        }
    }
}

// ---------------------------------------------------------------------------
// Kernel 2: flash attention.  Block = 8 waves, one (b,h), 128 q rows.
// K/V chunks of 64 kv positions staged in LDS by the Tensor Data Mover,
// double-buffered: wave 0 issues TDM for chunk c+1 while all waves compute
// chunk c from LDS; s_wait_tensorcnt + barrier closes the pipeline stage.
// Scores computed TRANSPOSED (S^T = K_tile x Q^T) so softmax stats need one
// shfl_xor(16) and exp'd probs land lane-local in the P.V A-fragment layout.
// ---------------------------------------------------------------------------
__global__ __launch_bounds__(256) void flash_attn_kernel(
    const _Float16* __restrict__ Q, const _Float16* __restrict__ K,
    const _Float16* __restrict__ Vt, _Float16* __restrict__ AO)
{
    // [buf][ K: 64kv x 64hd | V: 64d x 64kv ] halves; 16 KB per buffer
    __shared__ _Float16 smem[2 * 8192];

    const int bh    = blockIdx.x >> 5;
    const int qgrp  = blockIdx.x & 31;
    const int wid   = threadIdx.x >> 5;
    const int qbase = (qgrp * 8 + wid) * 16;
    const int lane  = threadIdx.x & 31;
    const int lo    = lane & 15;
    const int hi    = lane >> 4;

    const _Float16* Qh  = Q  + (size_t)bh * S_LEN * HD;
    const _Float16* Kh  = K  + (size_t)bh * S_LEN * HD;
    const _Float16* Vth = Vt + (size_t)bh * HD * S_LEN;
    const unsigned  smem_lds = (unsigned)(size_t)(const void*)&smem[0];

    // Q^T B-fragments (hd chunks [0,32) and [32,64)): load as Q rows.
    H16 qf[2];
    const _Float16* qrow = Qh + (size_t)(qbase + lo) * HD + hi * 8;
    qf[0].h[0] = *(const v8h*)(qrow);
    qf[0].h[1] = *(const v8h*)(qrow + 16);
    qf[1].h[0] = *(const v8h*)(qrow + 32);
    qf[1].h[1] = *(const v8h*)(qrow + 48);

    v8f acc[4] = {};
    float m = -1e30f, l = 0.0f;

    constexpr int NCHUNK = S_LEN / 64;     // 64 chunks of 64 kv

    // Prologue: stage chunk 0 into buffer 0.
    if (wid == 0) {
        tdm_load_2d(smem_lds,             Kh,  64, 64, 64, 64, 64);        // K: 64 rows x 64
        tdm_load_2d(smem_lds + 8192,      Vth, 64, 64, 64, 64, S_LEN);     // V^T: 64 d-rows x 64 kv
        __builtin_amdgcn_s_wait_tensorcnt(0);
    }
    __syncthreads();

    for (int c = 0; c < NCHUNK; ++c) {
        const int cur = c & 1;
        if (wid == 0 && (c + 1) < NCHUNK) {
            const int nkv = (c + 1) * 64;
            const unsigned dst = smem_lds + (cur ^ 1) * 16384;
            tdm_load_2d(dst,        Kh + (size_t)nkv * HD, 64, 64, 64, 64, 64);
            tdm_load_2d(dst + 8192, Vth + nkv,             64, 64, 64, 64, S_LEN);
        }
        const _Float16* Kc = smem + cur * 8192;           // [kv][hd]
        const _Float16* Vc = Kc + 4096;                   // [d][kv]

#pragma unroll
        for (int s2 = 0; s2 < 2; ++s2) {                  // two 32-kv sub-chunks
            // K A-fragments from LDS: two 16-row kv tiles x two hd chunks
            H16 kf[2][2];
#pragma unroll
            for (int t = 0; t < 2; ++t) {
                const _Float16* krow = Kc + (size_t)(s2 * 32 + t * 16 + lo) * 64 + hi * 8;
                kf[t][0].h[0] = *(const v8h*)(krow);
                kf[t][0].h[1] = *(const v8h*)(krow + 16);
                kf[t][1].h[0] = *(const v8h*)(krow + 32);
                kf[t][1].h[1] = *(const v8h*)(krow + 48);
            }
            // V B-fragments from LDS (transposed V): contiguous along kv
            H16 vf[4];
#pragma unroll
            for (int t = 0; t < 4; ++t) {
                const _Float16* vrow = Vc + (size_t)(t * 16 + lo) * 64 + s2 * 32 + hi * 8;
                vf[t].h[0] = *(const v8h*)(vrow);
                vf[t].h[1] = *(const v8h*)(vrow + 16);
            }

            v8f st0 = {}, st1 = {};
            st0 = WMMA_F16(kf[0][0].v, qf[0].v, st0);
            st0 = WMMA_F16(kf[0][1].v, qf[1].v, st0);
            st1 = WMMA_F16(kf[1][0].v, qf[0].v, st1);
            st1 = WMMA_F16(kf[1][1].v, qf[1].v, st1);

            // Online softmax; per-lane stats keyed by q = lane&15.
            float cmax = -1e30f;
#pragma unroll
            for (int r = 0; r < 8; ++r)
                cmax = fmaxf(cmax, fmaxf(st0[r], st1[r]));
            cmax = fmaxf(cmax, __shfl_xor(cmax, 16, 32));
            const float mnew  = fmaxf(m, cmax);
            const float alpha = __builtin_amdgcn_exp2f((m - mnew) * KSCALE);
            const float mk    = mnew * KSCALE;

            H16 pf;
            float rsum = 0.0f;
#pragma unroll
            for (int r = 0; r < 8; ++r) {
                const float p0 = __builtin_amdgcn_exp2f(st0[r] * KSCALE - mk);
                const float p1 = __builtin_amdgcn_exp2f(st1[r] * KSCALE - mk);
                rsum += p0 + p1;
                pf.v[r]     = (_Float16)p0;
                pf.v[r + 8] = (_Float16)p1;
            }
            rsum += __shfl_xor(rsum, 16, 32);
            l = l * alpha + rsum;
            m = mnew;

            float av[8];
#pragma unroll
            for (int r = 0; r < 8; ++r)
                av[r] = __shfl(alpha, r + 8 * hi, 32);

#pragma unroll
            for (int t = 0; t < 4; ++t) {
#pragma unroll
                for (int r = 0; r < 8; ++r) acc[t][r] *= av[r];
                acc[t] = WMMA_F16(pf.v, vf[t].v, acc[t]);
            }
        }

        if (wid == 0) __builtin_amdgcn_s_wait_tensorcnt(0);
        __syncthreads();
    }

    // Normalize and store f16 into AO [B, S, 256]
    const float linv = 1.0f / l;
    float lv[8];
#pragma unroll
    for (int r = 0; r < 8; ++r)
        lv[r] = __shfl(linv, r + 8 * hi, 32);

    const int b = bh >> 2, h = bh & 3;
    _Float16* aor = AO + ((size_t)b * S_LEN + qbase + 8 * hi) * DIM + h * HD + lo;
#pragma unroll
    for (int t = 0; t < 4; ++t)
#pragma unroll
        for (int r = 0; r < 8; ++r)
            aor[(size_t)r * DIM + t * 16] = (_Float16)(acc[t][r] * lv[r]);
}

// ---------------------------------------------------------------------------
// Kernel 3: output projection.  out = AO @ wo.T + bo  (f32 out)
// ---------------------------------------------------------------------------
__global__ __launch_bounds__(256) void out_proj_kernel(
    const _Float16* __restrict__ AO, const float* __restrict__ wo,
    const float* __restrict__ bo, float* __restrict__ out)
{
    const int w    = blockIdx.x * 8 + (threadIdx.x >> 5);
    const int mb   = (w >> 4) * 16;
    const int nb   = (w & 15) * 16;
    const int lane = threadIdx.x & 31;
    const int lo   = lane & 15;
    const int hi   = lane >> 4;

    v8f acc = {};
    const _Float16* ar = AO + (size_t)(mb + lo) * DIM + hi * 8;
    const float*    wr = wo + (size_t)(nb + lo) * DIM + hi * 8;
#pragma unroll
    for (int k = 0; k < DIM; k += 32) {
        H16 af, bf;
        af.h[0] = *(const v8h*)(ar + k);
        af.h[1] = *(const v8h*)(ar + k + 16);
        v8f b0 = *(const v8f*)(wr + k);
        v8f b1 = *(const v8f*)(wr + k + 16);
#pragma unroll
        for (int i = 0; i < 8; ++i) {
            bf.v[i]     = (_Float16)b0[i];
            bf.v[i + 8] = (_Float16)b1[i];
        }
        acc = WMMA_F16(af.v, bf.v, acc);
    }

    const float bias = bo[nb + lo];
#pragma unroll
    for (int r = 0; r < 8; ++r)
        out[(size_t)(mb + 8 * hi + r) * DIM + nb + lo] = acc[r] + bias;
}

// ---------------------------------------------------------------------------
extern "C" void kernel_launch(void* const* d_in, const int* in_sizes, int n_in,
                              void* d_out, int out_size, void* d_ws, size_t ws_size,
                              hipStream_t stream) {
    (void)in_sizes; (void)n_in; (void)out_size; (void)ws_size;
    const float* x  = (const float*)d_in[0];
    const float* wq = (const float*)d_in[1];
    const float* wk = (const float*)d_in[2];
    const float* wv = (const float*)d_in[3];
    const float* wo = (const float*)d_in[4];
    const float* bo = (const float*)d_in[5];
    float* out = (float*)d_out;

    const size_t elems = (size_t)4 * S_LEN * DIM;   // 4,194,304 per tensor
    _Float16* Q  = (_Float16*)d_ws;
    _Float16* K  = Q  + elems;
    _Float16* Vt = K  + elems;
    _Float16* AO = Vt + elems;

    qkv_proj_kernel<<<6144, 256, 0, stream>>>(x, wq, wk, wv, Q, K, Vt);
    flash_attn_kernel<<<512, 256, 0, stream>>>(Q, K, Vt, AO);
    out_proj_kernel<<<2048, 256, 0, stream>>>(AO, wo, bo, out);
}
